// EPSparseMoE_70360154243384
// MI455X (gfx1250) — compile-verified, matrix operand order checked by reading, not run on hardware
//
#include <hip/hip_runtime.h>
#include <hip/hip_bf16.h>
#include <stdint.h>

// ---------------------------------------------------------------------------
// EP Sparse MoE forward for MI455X (gfx1250, wave32, WMMA).
// Expert GEMMs: bf16 WMMA (v_wmma_f32_16x16x32_bf16), f32 accumulate,
// double-buffered LDS with async global->LDS staging for the activation tile.
// ---------------------------------------------------------------------------

#define Bb 4
#define Ll 1024
#define Dd 1024
#define Ee 8
#define Hh 4096
#define Nn (Bb * Ll)   // 4096 tokens

typedef __attribute__((ext_vector_type(16))) __bf16 v16bf;
typedef __attribute__((ext_vector_type(8)))  float  v8f;
typedef int v4i_ __attribute__((vector_size(16)));

union Frag {
    v16bf        v;
    unsigned int u[8];
    uint4        q[2];
};

#if defined(__has_builtin)
#if __has_builtin(__builtin_amdgcn_global_load_async_to_lds_b128)
#define MOE_HAVE_ASYNC 1
#endif
#if __has_builtin(__builtin_amdgcn_s_wait_asynccnt)
#define MOE_HAVE_WAIT_ASYNC 1
#endif
#endif

#ifdef MOE_HAVE_ASYNC
typedef __attribute__((address_space(1))) v4i_* as1_v4i_ptr;
typedef __attribute__((address_space(3))) v4i_* as3_v4i_ptr;
#endif

// Copy 16B global (bf16 data) -> 16B LDS. Async (ASYNCcnt) when available.
__device__ __forceinline__ void copy_b128_g2l(const __bf16* g, __bf16* l) {
#ifdef MOE_HAVE_ASYNC
    __builtin_amdgcn_global_load_async_to_lds_b128(
        (as1_v4i_ptr)(uintptr_t)g,
        (as3_v4i_ptr)(unsigned int)(uintptr_t)l,
        0, 0);
#else
    *(uint4*)l = *(const uint4*)g;
#endif
}

__device__ __forceinline__ void wait_async_copies() {
#ifdef MOE_HAVE_ASYNC
#ifdef MOE_HAVE_WAIT_ASYNC
    __builtin_amdgcn_s_wait_asynccnt(0);
#else
    asm volatile("s_wait_asynccnt 0x0" ::: "memory");
#endif
#endif
}

__device__ __forceinline__ float gelu_tanh(float x) {
    // jax.nn.gelu default (tanh approximation)
    float x3 = x * x * x;
    float t  = 0.7978845608028654f * (x + 0.044715f * x3);
    return 0.5f * x * (1.0f + tanhf(t));
}

// ---------------------------------------------------------------------------
// Kernel 0: convert x (fp32) -> bf16 workspace copy; zero per-expert counters.
// ---------------------------------------------------------------------------
__global__ void moe_prep_kernel(const float* __restrict__ x,
                                __bf16* __restrict__ xb,
                                int* __restrict__ cnt, int nd) {
    int i = blockIdx.x * 256 + threadIdx.x;
    if (i < nd) xb[i] = (__bf16)x[i];
    if (blockIdx.x == 0 && threadIdx.x < Ee) cnt[threadIdx.x] = 0;
}

// ---------------------------------------------------------------------------
// Kernel 1: router. One wave32 per token.
// ---------------------------------------------------------------------------
__global__ void __launch_bounds__(256)
moe_router_kernel(const float* __restrict__ x, const float* __restrict__ Wg,
                  const float* __restrict__ bg, float* __restrict__ logits_out,
                  int* __restrict__ cnt, int* __restrict__ tok,
                  int* __restrict__ slot, float* __restrict__ gate) {
    int wave = threadIdx.x >> 5;
    int lane = threadIdx.x & 31;
    int n = blockIdx.x * 8 + wave;

    float acc[Ee];
#pragma unroll
    for (int e = 0; e < Ee; ++e) acc[e] = 0.0f;

    const float* xr = x + (size_t)n * Dd;
    for (int d = lane; d < Dd; d += 32) {
        float xv = xr[d];
        const float* wr = Wg + (size_t)d * Ee;
#pragma unroll
        for (int e = 0; e < Ee; ++e) acc[e] = fmaf(xv, wr[e], acc[e]);
    }
#pragma unroll
    for (int e = 0; e < Ee; ++e) {
        for (int off = 16; off > 0; off >>= 1)
            acc[e] += __shfl_xor(acc[e], off, 32);
    }

    if (lane == 0) {
        float lg[Ee];
#pragma unroll
        for (int e = 0; e < Ee; ++e) {
            lg[e] = acc[e] + bg[e];
            logits_out[(size_t)n * Ee + e] = lg[e];
        }
        int bi = 0;
        float bv = lg[0];
#pragma unroll
        for (int e = 1; e < Ee; ++e)
            if (lg[e] > bv) { bv = lg[e]; bi = e; }
        int si = (bi == 0) ? 1 : 0;
        float sv = lg[si];
#pragma unroll
        for (int e = 0; e < Ee; ++e)
            if (e != bi && lg[e] > sv) { sv = lg[e]; si = e; }
        float w0 = 1.0f / (1.0f + expf(sv - bv));
        float w1 = 1.0f - w0;
        int p0 = atomicAdd(cnt + bi, 1);
        tok[bi * Nn + p0] = n; slot[bi * Nn + p0] = 0; gate[bi * Nn + p0] = w0;
        int p1 = atomicAdd(cnt + si, 1);
        tok[si * Nn + p1] = n; slot[si * Nn + p1] = 1; gate[si * Nn + p1] = w1;
    }
}

// ---------------------------------------------------------------------------
// GEMM tile machinery (128x128 tile / 256 threads / Kblk=32, double buffered).
//
// A tile is stored FRAGMENT-READY: for row-subtile mt (16 rows), lane L holds
// 16 contiguous bf16 (32B) at As[mt*512 + L*16]. Per ISA 7.12.2 the 16-bit
// A-fragment for lane L covers row M=L&15 with K = {0-7,16-23} (L<16) or
// {8-15,24-31} (L>=16). A 16B global chunk (row m, 8 consecutive K starting
// at g*8, g=0..3) therefore lands contiguously at
//   dst = As + (m>>4)*512 + ((m&15) + (g&1)*16)*16 + (g>>1)*8
// which is exactly what GLOBAL_LOAD_ASYNC_TO_LDS_B128 moves per lane.
// Fragment load = two ds_load_b128, no address arithmetic.
//
// B tile (fp32 -> bf16 conversion through VGPRs) keeps the conflict-free
// stride-34 bf16 layout; fragments are gathered as 8 dwords per lane.
// ---------------------------------------------------------------------------

#define AS_TILE (128 * 32)   // bf16 elements, fragment-ready
#define BS_TILE (128 * 34)   // bf16 elements, stride-34

__device__ __forceinline__ void issue_b_tile(const float* __restrict__ wsrc,
                                             int ldb, int kb,
                                             __bf16* __restrict__ Bst, int t) {
    int kr = t >> 5;          // 0..7
    int c4 = (t & 31) * 4;    // 0..124
#pragma unroll
    for (int p = 0; p < 4; ++p) {
        int k = p * 8 + kr;
        const float4 v = *(const float4*)(wsrc + (size_t)(kb + k) * ldb + c4);
        Bst[(c4 + 0) * 34 + k] = (__bf16)v.x;
        Bst[(c4 + 1) * 34 + k] = (__bf16)v.y;
        Bst[(c4 + 2) * 34 + k] = (__bf16)v.z;
        Bst[(c4 + 3) * 34 + k] = (__bf16)v.w;
    }
}

__device__ __forceinline__ int a_chunk_dst(int m, int g) {
    // bf16 index inside the fragment-ready A tile for chunk (row m, group g)
    return (m >> 4) * 512 + (((m & 15) + ((g & 1) << 4)) << 4) + ((g >> 1) << 3);
}

// FFN1: h[r,:] = gelu( x[tok[r],:] @ w1[e] + b1[e] )   (bf16 out)
__global__ void __launch_bounds__(256)
moe_ffn1_kernel(int e, const __bf16* __restrict__ xb,
                const int* __restrict__ tok, const int* __restrict__ cnt,
                const float* __restrict__ w1, const float* __restrict__ b1,
                __bf16* __restrict__ hbuf) {
    const int m0 = blockIdx.y * 128;
    const int c  = cnt[e];
    if (m0 >= c) return;
    const int nh0 = blockIdx.x * 128;

    __shared__ __align__(16) __bf16 As[2][AS_TILE];
    __shared__ __bf16 Bs[2][BS_TILE];
    __shared__ int    rows[128];

    const int t = threadIdx.x;
    if (t < 128) {
        int r = m0 + t;
        rows[t] = (r < c) ? tok[e * Nn + r] : tok[e * Nn];  // clamp to valid row
    }
    __syncthreads();

    const int wave = t >> 5, lane = t & 31;
    const int wy = wave >> 2, wx = wave & 3;   // 2 (M) x 4 (N) wave grid
    const int nBase = wx * 32;
    const int lm = lane & 15, lh = lane >> 4;

    v8f acc[4][2] = {};
    const float* wptr = w1 + (size_t)e * Dd * Hh + nh0;

    // ---- preload k-block 0 into buffer 0 ----
    {
#pragma unroll
        for (int pass = 0; pass < 2; ++pass) {
            int ch = pass * 256 + t;              // 512 chunks of 16B
            int m = ch >> 2, g = ch & 3;
            copy_b128_g2l(xb + (size_t)rows[m] * Dd + (g << 3),
                          &As[0][a_chunk_dst(m, g)]);
        }
        issue_b_tile(wptr, Hh, 0, Bs[0], t);
    }

    int cur = 0;
    for (int kb = 0; kb < Dd; kb += 32) {
        wait_async_copies();
        __syncthreads();

        const int nkb = kb + 32;
        if (nkb < Dd) {
#pragma unroll
            for (int pass = 0; pass < 2; ++pass) {
                int ch = pass * 256 + t;
                int m = ch >> 2, g = ch & 3;
                copy_b128_g2l(xb + (size_t)rows[m] * Dd + nkb + (g << 3),
                              &As[cur ^ 1][a_chunk_dst(m, g)]);
            }
            issue_b_tile(wptr, Hh, nkb, Bs[cur ^ 1], t);
        }

        Frag a[4], b[2];
#pragma unroll
        for (int mt = 0; mt < 4; ++mt) {
            const uint4* s =
                (const uint4*)&As[cur][(wy * 4 + mt) * 512 + lane * 16];
            a[mt].q[0] = s[0];
            a[mt].q[1] = s[1];
        }
#pragma unroll
        for (int nt = 0; nt < 2; ++nt) {
            const unsigned int* s =
                (const unsigned int*)Bs[cur] + (nBase + nt * 16 + lm) * 17 + lh * 4;
#pragma unroll
            for (int p = 0; p < 8; ++p) b[nt].u[p] = s[(p & 3) + ((p >> 2) * 8)];
        }
#pragma unroll
        for (int mt = 0; mt < 4; ++mt)
#pragma unroll
            for (int nt = 0; nt < 2; ++nt)
                acc[mt][nt] = __builtin_amdgcn_wmma_f32_16x16x32_bf16(
                    false, a[mt].v, false, b[nt].v, (short)0, acc[mt][nt],
                    false, false);
        cur ^= 1;
    }

    // epilogue: +b1, gelu, bf16 store (rows always < Nn; masking done in FFN2)
#pragma unroll
    for (int nt = 0; nt < 2; ++nt) {
        int colg = nh0 + nBase + nt * 16 + lm;
        float bias = b1[e * Hh + colg];
#pragma unroll
        for (int mt = 0; mt < 4; ++mt) {
#pragma unroll
            for (int j = 0; j < 8; ++j) {
                int row = m0 + wy * 64 + mt * 16 + lh * 8 + j;
                float v = acc[mt][nt][j] + bias;
                hbuf[(size_t)row * Hh + colg] = (__bf16)gelu_tanh(v);
            }
        }
    }
}

// FFN2: y[r,:] = h[r,:] @ w2[e] + b2[e]; scatter gate*y into ybuf[slot][token].
__global__ void __launch_bounds__(256)
moe_ffn2_kernel(int e, const __bf16* __restrict__ hbuf,
                const int* __restrict__ tok, const int* __restrict__ slot,
                const float* __restrict__ gate, const int* __restrict__ cnt,
                const float* __restrict__ w2, const float* __restrict__ b2,
                float* __restrict__ ybuf) {
    const int m0 = blockIdx.y * 128;
    const int c  = cnt[e];
    if (m0 >= c) return;
    const int d0 = blockIdx.x * 128;

    __shared__ __align__(16) __bf16 As[2][AS_TILE];
    __shared__ __bf16 Bs[2][BS_TILE];

    const int t = threadIdx.x;
    const int wave = t >> 5, lane = t & 31;
    const int wy = wave >> 2, wx = wave & 3;
    const int nBase = wx * 32;
    const int lm = lane & 15, lh = lane >> 4;

    v8f acc[4][2] = {};
    const float* wptr = w2 + (size_t)e * Hh * Dd + d0;

    {
#pragma unroll
        for (int pass = 0; pass < 2; ++pass) {
            int ch = pass * 256 + t;
            int m = ch >> 2, g = ch & 3;
            copy_b128_g2l(hbuf + (size_t)(m0 + m) * Hh + (g << 3),
                          &As[0][a_chunk_dst(m, g)]);
        }
        issue_b_tile(wptr, Dd, 0, Bs[0], t);
    }

    int cur = 0;
    for (int kb = 0; kb < Hh; kb += 32) {
        wait_async_copies();
        __syncthreads();

        const int nkb = kb + 32;
        if (nkb < Hh) {
#pragma unroll
            for (int pass = 0; pass < 2; ++pass) {
                int ch = pass * 256 + t;
                int m = ch >> 2, g = ch & 3;
                copy_b128_g2l(hbuf + (size_t)(m0 + m) * Hh + nkb + (g << 3),
                              &As[cur ^ 1][a_chunk_dst(m, g)]);
            }
            issue_b_tile(wptr, Dd, nkb, Bs[cur ^ 1], t);
        }

        Frag a[4], b[2];
#pragma unroll
        for (int mt = 0; mt < 4; ++mt) {
            const uint4* s =
                (const uint4*)&As[cur][(wy * 4 + mt) * 512 + lane * 16];
            a[mt].q[0] = s[0];
            a[mt].q[1] = s[1];
        }
#pragma unroll
        for (int nt = 0; nt < 2; ++nt) {
            const unsigned int* s =
                (const unsigned int*)Bs[cur] + (nBase + nt * 16 + lm) * 17 + lh * 4;
#pragma unroll
            for (int p = 0; p < 8; ++p) b[nt].u[p] = s[(p & 3) + ((p >> 2) * 8)];
        }
#pragma unroll
        for (int mt = 0; mt < 4; ++mt)
#pragma unroll
            for (int nt = 0; nt < 2; ++nt)
                acc[mt][nt] = __builtin_amdgcn_wmma_f32_16x16x32_bf16(
                    false, a[mt].v, false, b[nt].v, (short)0, acc[mt][nt],
                    false, false);
        cur ^= 1;
    }

    // epilogue: scatter gate*(y+b2) into per-slot buffer (each (token,slot) unique)
#pragma unroll
    for (int nt = 0; nt < 2; ++nt) {
        int col = d0 + nBase + nt * 16 + lm;
        float bias = b2[e * Dd + col];
#pragma unroll
        for (int mt = 0; mt < 4; ++mt) {
#pragma unroll
            for (int j = 0; j < 8; ++j) {
                int r = m0 + wy * 64 + mt * 16 + lh * 8 + j;
                if (r < c) {
                    int   n = tok[e * Nn + r];
                    int   s = slot[e * Nn + r];
                    float g = gate[e * Nn + r];
                    ybuf[(size_t)s * Nn * Dd + (size_t)n * Dd + col] =
                        g * (acc[mt][nt][j] + bias);
                }
            }
        }
    }
}

// ---------------------------------------------------------------------------
// Kernel 4: out = ybuf[slot0] + ybuf[slot1]
// ---------------------------------------------------------------------------
__global__ void moe_combine_kernel(const float* __restrict__ ybuf,
                                   float* __restrict__ out, int nd) {
    int i = blockIdx.x * 256 + threadIdx.x;
    if (i < nd) out[i] = ybuf[i] + ybuf[(size_t)nd + i];
}

// ---------------------------------------------------------------------------
extern "C" void kernel_launch(void* const* d_in, const int* in_sizes, int n_in,
                              void* d_out, int out_size, void* d_ws, size_t ws_size,
                              hipStream_t stream) {
    const float* x  = (const float*)d_in[0];
    const float* Wg = (const float*)d_in[1];
    const float* bg = (const float*)d_in[2];
    const float* w1 = (const float*)d_in[3];
    const float* b1 = (const float*)d_in[4];
    const float* w2 = (const float*)d_in[5];
    const float* b2 = (const float*)d_in[6];

    float* out    = (float*)d_out;
    float* logits = out + (size_t)Nn * Dd;   // tuple output #2

    char* ws = (char*)d_ws;
    size_t o = 0;
    int*    cnt  = (int*)(ws + o);    o += 256;
    int*    tok  = (int*)(ws + o);    o += (size_t)Ee * Nn * 4;
    int*    slt  = (int*)(ws + o);    o += (size_t)Ee * Nn * 4;
    float*  gate = (float*)(ws + o);  o += (size_t)Ee * Nn * 4;
    __bf16* xb   = (__bf16*)(ws + o); o += (size_t)Nn * Dd * 2;
    __bf16* hbuf = (__bf16*)(ws + o); o += (size_t)Nn * Hh * 2;
    float*  ybuf = (float*)(ws + o);  o += (size_t)2 * Nn * Dd * 4;
    (void)ws_size; (void)in_sizes; (void)n_in; (void)out_size;

    const int nd = Nn * Dd;

    moe_prep_kernel<<<(nd + 255) / 256, 256, 0, stream>>>(x, xb, cnt, nd);
    moe_router_kernel<<<Nn / 8, 256, 0, stream>>>(x, Wg, bg, logits,
                                                  cnt, tok, slt, gate);
    for (int e = 0; e < Ee; ++e) {
        moe_ffn1_kernel<<<dim3(Hh / 128, Nn / 128), 256, 0, stream>>>(
            e, xb, tok, cnt, w1, b1, hbuf);
        moe_ffn2_kernel<<<dim3(Dd / 128, Nn / 128), 256, 0, stream>>>(
            e, hbuf, tok, slt, gate, cnt, w2, b2, ybuf);
    }
    moe_combine_kernel<<<(nd + 255) / 256, 256, 0, stream>>>(ybuf, out, nd);
}
